// CausalSelfAttention_41025527611821
// MI455X (gfx1250) — compile-verified, matrix operand order checked by reading, not run on hardware
//
#include <hip/hip_runtime.h>

// ---------------------------------------------------------------------------
// CDNA5 (gfx1250) wave32 WMMA implementation of causal self-attention.
// All matmuls use v_wmma_f32_16x16x32_bf16 (fp32 accumulate).
// Double-buffered LDS with fetch/compute/commit software pipeline.
// ---------------------------------------------------------------------------

typedef __bf16 bf16;
typedef __attribute__((ext_vector_type(16))) __bf16 v16bf;
typedef __attribute__((ext_vector_type(8)))  __bf16 bf16x8;
typedef __attribute__((ext_vector_type(4)))  __bf16 bf16x4;
typedef __attribute__((ext_vector_type(8)))  float  v8f;
typedef __attribute__((ext_vector_type(4)))  float  f32x4;

// Load one 16x32 (or 32x16) bf16 WMMA operand fragment from LDS.
// Per ISA layout: lane half selects kbase (0 or 8); element j holds
// k = kbase + j (j<8) or kbase + j + 8 (j>=8)  => two contiguous 16B runs.
__device__ __forceinline__ v16bf load_frag(const bf16* row, int kb) {
    bf16x8 lo = *(const bf16x8*)(row + kb);        // k = kb .. kb+7
    bf16x8 hi = *(const bf16x8*)(row + kb + 16);   // k = kb+16 .. kb+23
    return __builtin_shufflevector(lo, hi, 0,1,2,3,4,5,6,7,8,9,10,11,12,13,14,15);
}

__device__ __forceinline__ v8f wmma_bf16(v16bf a, v16bf b, v8f c) {
    return __builtin_amdgcn_wmma_f32_16x16x32_bf16(false, a, false, b,
                                                   (short)0, c, false, false);
}

__device__ __forceinline__ bf16x4 cvt4(f32x4 v) {
    bf16x4 r;
    r[0] = (bf16)v[0]; r[1] = (bf16)v[1]; r[2] = (bf16)v[2]; r[3] = (bf16)v[3];
    return r;
}

// ---------------------------------------------------------------------------
// GEMM: C[M,N] = A[M,K] @ B[K,N] + bias[N]   (fp32 in/out, bf16 WMMA inside)
// Block tile 64(M) x 256(N), 256 threads = 8 waves (2x4), 32x64 per wave.
// 8 WMMAs per wave per 32-deep K step; double-buffered LDS pipeline.
// ---------------------------------------------------------------------------
#define GK 32
#define GPAD 8
#define GROW (GK + GPAD)

__global__ __launch_bounds__(256)
void gemm_bias_bf16(const float* __restrict__ A, const float* __restrict__ B,
                    const float* __restrict__ bias, float* __restrict__ C,
                    int M, int N, int K) {
    __shared__ bf16 As[2][64][GROW];     // [m][k]
    __shared__ bf16 Bs[2][256][GROW];    // [n][k] (transposed at stage-in)

    const int tid  = threadIdx.x;
    const int wave = tid >> 5;
    const int lane = tid & 31;
    const int wm = (wave >> 2) * 32;      // wave row offset (0/32)
    const int wn = (wave & 3) * 64;       // wave col offset (0..192)
    const int bm = blockIdx.y * 64;
    const int bn = blockIdx.x * 256;

    const int row = lane & 15;
    const int kb  = 8 * (lane >> 4);

    // Per-thread staging coordinates
    const int ar  = tid >> 3;             // A row 0..31 (+32 for 2nd half)
    const int ac4 = (tid & 7) * 4;        // A col 0..28
    const int br  = tid >> 6;             // B k-row 0..3 (+4*it)
    const int bc4 = (tid & 63) * 4;       // B n-col 0..252

    f32x4 fa0, fa1, fb[8];                // fetch registers

    auto fetch = [&](int k0) {
        const float* aptr = &A[(long long)(bm + ar) * K + k0 + ac4];
        fa0 = *(const f32x4*)aptr;
        fa1 = *(const f32x4*)(aptr + 32LL * K);
        const float* bptr = &B[(long long)(k0 + br) * N + bn + bc4];
        #pragma unroll
        for (int it = 0; it < 8; ++it)
            fb[it] = *(const f32x4*)(bptr + (long long)(it * 4) * N);
        // speculative prefetch two tiles ahead (silently dropped if OOB)
        __builtin_prefetch(aptr + 2 * GK, 0, 3);
        __builtin_prefetch(bptr + (long long)(2 * GK) * N, 0, 3);
    };
    auto commit = [&](int buf) {
        *(bf16x4*)&As[buf][ar][ac4]      = cvt4(fa0);
        *(bf16x4*)&As[buf][ar + 32][ac4] = cvt4(fa1);
        #pragma unroll
        for (int it = 0; it < 8; ++it) {
            bf16x4 w = cvt4(fb[it]);
            int r = br + it * 4;
            Bs[buf][bc4 + 0][r] = w[0];
            Bs[buf][bc4 + 1][r] = w[1];
            Bs[buf][bc4 + 2][r] = w[2];
            Bs[buf][bc4 + 3][r] = w[3];
        }
    };

    v8f acc[2][4] = {};
    const int nk = K / GK;

    fetch(0);
    commit(0);
    __syncthreads();

    for (int kk = 0; kk < nk; ++kk) {
        const int buf = kk & 1;
        if (kk + 1 < nk) fetch((kk + 1) * GK);   // global loads issue early

        v16bf af[2], bfrag[4];
        #pragma unroll
        for (int i = 0; i < 2; ++i)
            af[i] = load_frag(&As[buf][wm + i * 16 + row][0], kb);
        #pragma unroll
        for (int j = 0; j < 4; ++j)
            bfrag[j] = load_frag(&Bs[buf][wn + j * 16 + row][0], kb);

        #pragma unroll
        for (int i = 0; i < 2; ++i)
            #pragma unroll
            for (int j = 0; j < 4; ++j)
                acc[i][j] = wmma_bf16(af[i], bfrag[j], acc[i][j]);

        if (kk + 1 < nk) commit(buf ^ 1);        // convert + LDS store
        __syncthreads();
    }

    // C/D layout: lane 0-15 -> n=lane, m=r ; lane 16-31 -> n=lane-16, m=r+8
    const int n_lane = lane & 15;
    const int m_hi   = (lane >> 4) * 8;
    #pragma unroll
    for (int i = 0; i < 2; ++i)
        #pragma unroll
        for (int j = 0; j < 4; ++j) {
            int col = bn + wn + j * 16 + n_lane;
            float bv = bias[col];
            #pragma unroll
            for (int r = 0; r < 8; ++r) {
                int rrow = bm + wm + i * 16 + m_hi + r;
                C[(long long)rrow * N + col] = acc[i][j][r] + bv;
            }
        }
}

// ---------------------------------------------------------------------------
// Flash attention, one workgroup per (b, h, 64-row query block).
// 128 threads = 4 waves, 16 query rows per wave. dk = 64, key block = 64.
// QKV layout: [B*T, 3*1024] with Q at col h*64, K at 1024+h*64, V at 2048+h*64.
// Double-buffered K/V tiles.
// ---------------------------------------------------------------------------
#define FPAD 8
#define FROW (64 + FPAD)

__global__ __launch_bounds__(128)
void flash_attn_wmma(const float* __restrict__ QKV, float* __restrict__ Y) {
    __shared__ bf16 Ks[2][64][FROW];     // [key][d]
    __shared__ bf16 Vs[2][64][FROW];     // [d][key]  (transposed)
    __shared__ bf16 Ps[4][16][FROW];     // per-wave P scratch [m][key]

    const int tid  = threadIdx.x;
    const int wave = tid >> 5;
    const int lane = tid & 31;
    const int bh = blockIdx.y;           // 0..31
    const int b  = bh >> 4;
    const int h  = bh & 15;
    const int qb0 = blockIdx.x * 64;
    const int qr0 = wave * 16;

    const long long rstride = 3072;
    const float* base = QKV + (long long)b * 2048 * rstride;

    const int row = lane & 15;
    const int kb  = 8 * (lane >> 4);
    const int m_hi = (lane >> 4) * 8;

    // Per-thread staging coordinates (8 float4 per matrix per thread)
    const int sr  = tid >> 4;            // key row 0..7 (+8*it)
    const int sc4 = (tid & 15) * 4;      // d 0..60
    f32x4 fk[8], fv[8];

    auto fetch = [&](int kb0) {
        #pragma unroll
        for (int it = 0; it < 8; ++it) {
            const float* kp = base + (long long)(kb0 + sr + it * 8) * rstride
                              + 1024 + h * 64 + sc4;
            fk[it] = *(const f32x4*)kp;
            fv[it] = *(const f32x4*)(kp + 1024);    // V is 1024 cols later
        }
    };
    auto commit = [&](int buf) {
        #pragma unroll
        for (int it = 0; it < 8; ++it) {
            int r = sr + it * 8;
            *(bf16x4*)&Ks[buf][r][sc4] = cvt4(fk[it]);
            bf16x4 w = cvt4(fv[it]);
            Vs[buf][sc4 + 0][r] = w[0];
            Vs[buf][sc4 + 1][r] = w[1];
            Vs[buf][sc4 + 2][r] = w[2];
            Vs[buf][sc4 + 3][r] = w[3];
        }
    };

    // Q fragments (A-operand), scaled by 1/sqrt(dk)=0.125, kept in registers.
    v16bf qf[2];
    {
        const float* qrow = base + (long long)(qb0 + qr0 + row) * rstride + h * 64;
        #pragma unroll
        for (int half = 0; half < 2; ++half) {
            const float* p = qrow + 32 * half + kb;
            f32x4 a0 = *(const f32x4*)(p);
            f32x4 a1 = *(const f32x4*)(p + 4);
            f32x4 a2 = *(const f32x4*)(p + 16);
            f32x4 a3 = *(const f32x4*)(p + 20);
            v16bf t;
            #pragma unroll
            for (int j = 0; j < 4; ++j) {
                t[j]      = (bf16)(a0[j] * 0.125f);
                t[4 + j]  = (bf16)(a1[j] * 0.125f);
                t[8 + j]  = (bf16)(a2[j] * 0.125f);
                t[12 + j] = (bf16)(a3[j] * 0.125f);
            }
            qf[half] = t;
        }
    }

    float m_row[8], l_row[8];
    v8f o[4] = {};
    #pragma unroll
    for (int r = 0; r < 8; ++r) { m_row[r] = -3.0e38f; l_row[r] = 0.f; }

    const int nblocks = blockIdx.x + 1;  // causal: keys up to qb0+63

    fetch(0);
    commit(0);
    __syncthreads();

    for (int jb = 0; jb < nblocks; ++jb) {
        const int kb0 = jb * 64;
        const int buf = jb & 1;
        if (jb + 1 < nblocks) fetch(kb0 + 64);   // next K/V global loads early

        // ---- S = Q @ K^T : 4 tiles of 16x16 per wave ----
        v8f s[4];
        #pragma unroll
        for (int t = 0; t < 4; ++t) {
            v16bf kf0 = load_frag(&Ks[buf][t * 16 + row][0], kb);   // d 0..31
            v16bf kf1 = load_frag(&Ks[buf][t * 16 + row][32], kb);  // d 32..63
            v8f a = {};
            a = wmma_bf16(qf[0], kf0, a);
            a = wmma_bf16(qf[1], kf1, a);
            s[t] = a;
        }

        // ---- Causal mask + running row max (rows split across lane halves) ----
        float mnew[8];
        #pragma unroll
        for (int r = 0; r < 8; ++r) mnew[r] = m_row[r];
        const int qbase = qb0 + qr0 + m_hi;
        #pragma unroll
        for (int t = 0; t < 4; ++t) {
            int keyg = kb0 + t * 16 + (lane & 15);
            #pragma unroll
            for (int r = 0; r < 8; ++r) {
                float v = s[t][r];
                if (keyg > qbase + r) v = -3.0e38f;
                s[t][r] = v;
                mnew[r] = fmaxf(mnew[r], v);
            }
        }
        #pragma unroll
        for (int r = 0; r < 8; ++r) {      // width-16 max reduce
            float v = mnew[r];
            #pragma unroll
            for (int off = 1; off < 16; off <<= 1)
                v = fmaxf(v, __shfl_xor(v, off, 32));
            mnew[r] = v;
        }

        // ---- exp + row sums ----
        float rsum[8] = {0.f, 0.f, 0.f, 0.f, 0.f, 0.f, 0.f, 0.f};
        #pragma unroll
        for (int t = 0; t < 4; ++t)
            #pragma unroll
            for (int r = 0; r < 8; ++r) {
                float p = __expf(s[t][r] - mnew[r]);
                s[t][r] = p;
                rsum[r] += p;
            }
        #pragma unroll
        for (int r = 0; r < 8; ++r) {      // width-16 sum reduce
            float v = rsum[r];
            #pragma unroll
            for (int off = 1; off < 16; off <<= 1)
                v += __shfl_xor(v, off, 32);
            rsum[r] = v;
        }

        // ---- Online rescale of l and O accumulators ----
        #pragma unroll
        for (int r = 0; r < 8; ++r) {
            float sc = __expf(m_row[r] - mnew[r]);
            l_row[r] = l_row[r] * sc + rsum[r];
            m_row[r] = mnew[r];
            #pragma unroll
            for (int t = 0; t < 4; ++t) o[t][r] *= sc;
        }

        // ---- C/D-layout P -> per-wave LDS -> reload in A-layout ----
        #pragma unroll
        for (int t = 0; t < 4; ++t)
            #pragma unroll
            for (int r = 0; r < 8; ++r)
                Ps[wave][m_hi + r][t * 16 + (lane & 15)] = (bf16)s[t][r];
        asm volatile("s_wait_dscnt 0" ::: "memory");  // per-wave LDS RAW fence

        v16bf pf0 = load_frag(&Ps[wave][row][0], kb);   // keys 0..31
        v16bf pf1 = load_frag(&Ps[wave][row][32], kb);  // keys 32..63
        #pragma unroll
        for (int t = 0; t < 4; ++t) {
            v16bf vf0 = load_frag(&Vs[buf][t * 16 + row][0], kb);
            v16bf vf1 = load_frag(&Vs[buf][t * 16 + row][32], kb);
            o[t] = wmma_bf16(pf0, vf0, o[t]);
            o[t] = wmma_bf16(pf1, vf1, o[t]);
        }

        if (jb + 1 < nblocks) commit(buf ^ 1);   // stage next block into LDS
        __syncthreads();
    }

    // ---- Normalize and store Y in [B,T,H*dk] = [4096,1024] ----
    #pragma unroll
    for (int t = 0; t < 4; ++t) {
        int d = t * 16 + (lane & 15);
        #pragma unroll
        for (int r = 0; r < 8; ++r) {
            int trow = qb0 + qr0 + m_hi + r;
            Y[(long long)(b * 2048 + trow) * 1024 + h * 64 + d] = o[t][r] / l_row[r];
        }
    }
}

// ---------------------------------------------------------------------------
extern "C" void kernel_launch(void* const* d_in, const int* in_sizes, int n_in,
                              void* d_out, int out_size, void* d_ws, size_t ws_size,
                              hipStream_t stream) {
    const float* x     = (const float*)d_in[0];  // [2,2048,1024]
    const float* qkv_w = (const float*)d_in[1];  // [1024,3072]
    const float* qkv_b = (const float*)d_in[2];  // [3072]
    const float* out_w = (const float*)d_in[3];  // [1024,1024]
    const float* out_b = (const float*)d_in[4];  // [1024]
    float* out = (float*)d_out;                  // [2,2048,1024]

    float* qkv = (float*)d_ws;                   // 4096*3072 fp32 (48 MB)
    float* y   = qkv + (size_t)4096 * 3072;      // 4096*1024 fp32 (16 MB)

    // 1) QKV projection: [4096,1024] @ [1024,3072] + bias
    gemm_bias_bf16<<<dim3(3072 / 256, 4096 / 64), dim3(256), 0, stream>>>(
        x, qkv_w, qkv_b, qkv, 4096, 3072, 1024);

    // 2) Causal flash attention per (b,h,query-block)
    flash_attn_wmma<<<dim3(2048 / 64, 32), dim3(128), 0, stream>>>(qkv, y);

    // 3) Output projection: [4096,1024] @ [1024,1024] + bias
    gemm_bias_bf16<<<dim3(1024 / 256, 4096 / 64), dim3(256), 0, stream>>>(
        y, out_w, out_b, out, 4096, 1024, 1024);
}
